// RecurrentDecoder_50242527428850
// MI455X (gfx1250) — matrix operand order repaired and checked by reading, not compiled, not run on hardware
//
#include <hip/hip_runtime.h>

typedef _Float16 h16_t;
typedef __attribute__((ext_vector_type(4)))  h16_t v4h;
typedef __attribute__((ext_vector_type(8)))  h16_t v8h;
typedef __attribute__((ext_vector_type(16))) h16_t v16h;
typedef __attribute__((ext_vector_type(8)))  float v8f;
typedef __attribute__((ext_vector_type(4)))  unsigned int v4u;
typedef __attribute__((ext_vector_type(4)))  int v4i;
typedef __attribute__((ext_vector_type(8)))  int v8i;

#define TD 64     // decoder steps
#define NB 32     // batch
#define EE 256    // encoder feature size
#define DD 512    // hidden size
#define G3 1536   // 3*D (GRU gates)
#define KX 512    // 2*E (GRU input size)
#define HK 768    // E + D (head input size)
#define NM 80     // n_mel
#define TE 256    // encoder time

// Tensor Data Mover availability (device pass only; host pass sees 0)
#if defined(__has_builtin)
#if __has_builtin(__builtin_amdgcn_tensor_load_to_lds) && \
    __has_builtin(__builtin_amdgcn_s_wait_tensorcnt)
#define USE_TDM 1
#endif
#endif
#ifndef USE_TDM
#define USE_TDM 0
#endif

// ---------------------------------------------------------------------------
// CDNA5 wave32 f16 WMMA fragment helpers (A: row-major MxK, B: row-major NxK).
//   A lane: row = lane&15; per 32-K step K-chunk  hs*16..+16
//   B lane: col = lane&15; per 32-K step K-chunk  hs*16..+16
//   C lane: elem r -> row r + 8*(lane>>4), col lane&15
// ---------------------------------------------------------------------------
__device__ __forceinline__ v16h ldA(const h16_t* __restrict__ p) {
  v8h a0 = *(const v8h*)p;
  v8h a1 = *(const v8h*)(p + 8);
  return __builtin_shufflevector(a0, a1, 0,1,2,3,4,5,6,7,8,9,10,11,12,13,14,15);
}
__device__ __forceinline__ v16h ldB(const h16_t* __restrict__ p) {
  v8h b0 = *(const v8h*)p;
  v8h b1 = *(const v8h*)(p + 8);
  return __builtin_shufflevector(b0, b1, 0,1,2,3,4,5,6,7,8,9,10,11,12,13,14,15);
}
__device__ __forceinline__ v8f wmma_step(v16h a, v16h b, v8f c) {
  return __builtin_amdgcn_wmma_f32_16x16x32_f16(false, a, false, b, (short)0, c,
                                                false, false);
}
__device__ __forceinline__ v8f wmma_mac16(const h16_t* __restrict__ A, int lda,
                                          const h16_t* __restrict__ B, int ldb,
                                          int K, int lane) {
  v8f acc = {0.f, 0.f, 0.f, 0.f, 0.f, 0.f, 0.f, 0.f};
  const int m  = lane & 15;
  const int hs = lane >> 4;
  const h16_t* ap = A + m * lda + hs * 16;
  const h16_t* bp = B + m * ldb + hs * 16;
  for (int k = 0; k < K; k += 32)
    acc = wmma_step(ldA(ap + k), ldB(bp + k), acc);
  return acc;
}

// ---------------------------------------------------------------------------
// One-time per call: cast weights / shifted decoder sequence to f16, zero h,
// and build the f16 per-batch transposed alignment image [b][e][t] so the
// per-step LDS staging is a pure contiguous-row copy (TDM-friendly).
// ---------------------------------------------------------------------------
__global__ void k_convert(const float* __restrict__ W_att, const float* __restrict__ W_ih,
                          const float* __restrict__ W_hh, const float* __restrict__ W_mel,
                          const float* __restrict__ W_gate, const float* __restrict__ dec,
                          const float* __restrict__ alignI,
                          h16_t* wa16, h16_t* wih16, h16_t* whh16, h16_t* whead16,
                          h16_t* seq16, h16_t* al16T, float* h32, h16_t* h16a) {
  const int stride = gridDim.x * blockDim.x;
  const int t0 = blockIdx.x * blockDim.x + threadIdx.x;
  for (int i = t0; i < EE * DD; i += stride) wa16[i]  = (h16_t)W_att[i];
  for (int i = t0; i < G3 * KX; i += stride) wih16[i] = (h16_t)W_ih[i];
  for (int i = t0; i < G3 * DD; i += stride) whh16[i] = (h16_t)W_hh[i];
  for (int i = t0; i < 96 * HK; i += stride) {            // [W_mel; W_gate; 0-pad]
    int r = i / HK, c = i - r * HK;
    float v = (r < NM) ? W_mel[r * HK + c] : ((r == NM) ? W_gate[c] : 0.f);
    whead16[i] = (h16_t)v;
  }
  for (int i = t0; i < TD * NB * EE; i += stride)         // teacher-forcing shift
    seq16[i] = (i < NB * EE) ? (h16_t)0.f : (h16_t)dec[i - NB * EE];
  for (int i = t0; i < NB * EE * TE; i += stride) {       // [b][e][t] <- (t,b,e)
    int b = i >> 16, rem = i & 65535;
    int e = rem >> 8, t = rem & 255;
    al16T[i] = (h16_t)alignI[(t * NB + b) * EE + e];
  }
  for (int i = t0; i < NB * DD; i += stride) { h32[i] = 0.f; h16a[i] = (h16_t)0.f; }
}

// ---------------------------------------------------------------------------
// Fused attention: one workgroup per batch element b.
//  - stage the f16 (E x T_enc) slice for b into LDS (TDM async DMA with
//    hardware row padding, overlapped with the proj WMMA; or copy fallback)
//  - proj row b via WMMA (8 waves x 2 tiles of h@W_att^T), row extracted to LDS
//  - phase A: per-e online softmax stats over t
//  - phase B: tw[t] = sum_e softmax weight
//  - phase C: ctx[e]; assemble x = [dec_in | ctx], dhac[:, :E] = ctx
// ---------------------------------------------------------------------------
#define ALD 260   // padded LDS row stride (halfs): 512B row + 8B pad
__global__ void k_attn(const h16_t* __restrict__ al16T, const h16_t* __restrict__ hcur,
                       const h16_t* __restrict__ wa16, const h16_t* __restrict__ seq16,
                       h16_t* __restrict__ x16, h16_t* __restrict__ dhac, int tstep) {
  extern __shared__ char smem_raw[];
  h16_t* al     = (h16_t*)smem_raw;                       // [e][ALD]
  float* proj_s = (float*)(smem_raw + TE * ALD * 2);
  float* m_s    = proj_s + TE;
  float* rz_s   = m_s + TE;
  float* tw_s   = rz_s + TE;
  const int b = blockIdx.x, tid = threadIdx.x;
  const int lane = tid & 31, wv = tid >> 5;
  const h16_t* src = al16T + (size_t)b * (TE * EE);       // 256 rows x 512 B

#if USE_TDM
  if (wv == 0) {
    // D# groups per CDNA5 ISA: 2D tensor, 4B elements,
    // tile 128dw x 256 rows, LDS pad 2 dwords every 128 dwords (-> ALD rows).
    const unsigned long long ga = (unsigned long long)(uintptr_t)src;
    const unsigned int lds0 = (unsigned int)__builtin_amdgcn_groupstaticsize();
    v4u g0 = { 1u,                                        // count=1, user mode
               lds0,                                      // lds_addr (bytes)
               (unsigned int)ga,                          // global_addr[31:0]
               (unsigned int)((ga >> 32) & 0x1FFFFFFu) | (2u << 30) }; // [56:32]|type=2
    v8i g1 = { (int)((2u << 16) | (1u << 20) | (6u << 22) | (1u << 25)),
               //    data_size=4B  pad_en      pad_int=128dw  pad_amt=2dw
               (int)(128u << 16),                         // tensor_dim0 = 128 dw
               (int)(256u << 16),                         // tensor_dim1 = 256 rows
               (int)(128u << 16),                         // tile_dim0 = 128 dw
               (int)256,                                  // tile_dim1 = 256 rows
               (int)128,                                  // tensor_dim0_stride = 128 dw
               0, 0 };                                    // dim1 stride unused (2D)
    v4i gz = {0, 0, 0, 0};
#if __clang_major__ >= 23
    v8i gz8 = {0, 0, 0, 0, 0, 0, 0, 0};
    __builtin_amdgcn_tensor_load_to_lds(g0, g1, gz, gz, gz8, 0);
#else
    __builtin_amdgcn_tensor_load_to_lds(g0, g1, gz, gz, 0);
#endif
  }
#else
  for (int idx = tid; idx < TE * EE / 4; idx += 256) {    // v4h copy fallback
    int e = idx >> 6, t4 = (idx & 63) << 2;
    *(v4h*)(al + e * ALD + t4) = *(const v4h*)(src + e * TE + t4);
  }
#endif

  { // proj row b = (h @ W_att^T)[b, :] via WMMA; overlaps the DMA above
    const int tm = b >> 4, lb = b & 15;
#pragma unroll
    for (int i = 0; i < 2; ++i) {
      const int tn = wv * 2 + i;
      v8f acc = wmma_mac16(hcur + tm * 16 * DD, DD, wa16 + tn * 16 * DD, DD, DD, lane);
      if ((lane >> 4) == (lb >> 3)) proj_s[tn * 16 + (lane & 15)] = acc[lb & 7];
    }
  }
#if USE_TDM
  if (wv == 0) __builtin_amdgcn_s_wait_tensorcnt(0);
#endif
  __syncthreads();
  { // phase A (tid = e): online softmax stats over t
    float p = proj_s[tid];
    float mx = -3.4e38f, Z = 0.f;
    for (int t = 0; t < TE; ++t) {
      float v = (float)al[tid * ALD + t] * p;
      if (v <= mx) Z += __expf(v - mx);
      else { Z = Z * __expf(mx - v) + 1.f; mx = v; }
    }
    m_s[tid] = mx; rz_s[tid] = 1.f / Z;
  }
  __syncthreads();
  { // phase B (tid = t): tw[t] = sum_e exp(s - m_e)/Z_e
    float s = 0.f;
    for (int e = 0; e < EE; ++e) {
      float v = (float)al[e * ALD + tid] * proj_s[e];
      s += __expf(v - m_s[e]) * rz_s[e];
    }
    tw_s[tid] = s;
  }
  __syncthreads();
  { // phase C (tid = e): ctx, write x = [dec_in | ctx] and dhac[:, :E] = ctx
    float c = 0.f;
    for (int t = 0; t < TE; ++t) c += tw_s[t] * (float)al[tid * ALD + t];
    x16[b * KX + EE + tid] = (h16_t)c;
    x16[b * KX + tid]      = seq16[(tstep * NB + b) * EE + tid];
    dhac[(tstep * NB + b) * HK + tid] = (h16_t)c;
  }
}

// ---------------------------------------------------------------------------
// Fused GRU: each wave owns a 16x16 (batch, hidden) patch and computes all six
// gate tiles (gi: r,z,n from x@W_ih^T; gh: r,z,n from h@W_hh^T) sharing the A
// fragments, then applies the gate math in registers and writes h directly.
// h16 is double-buffered across steps (gh reduction reads ALL of old h).
// ---------------------------------------------------------------------------
__global__ void k_gru(const h16_t* __restrict__ x16, const h16_t* __restrict__ hcur,
                      const h16_t* __restrict__ wih, const h16_t* __restrict__ whh,
                      float* __restrict__ h32, h16_t* __restrict__ hnext,
                      h16_t* __restrict__ dhac, int tstep) {
  const int lane = threadIdx.x & 31;
  const int w  = blockIdx.x * 4 + (threadIdx.x >> 5);     // 0..63
  const int tm = w >> 5, tn = w & 31;
  const int m  = lane & 15;
  const int hs = lane >> 4;

  const h16_t* ax = x16  + (tm * 16 + m) * KX + hs * 16;
  const h16_t* ah = hcur + (tm * 16 + m) * DD + hs * 16;
  const h16_t* bir = wih + (0 * DD + tn * 16 + m) * KX + hs * 16;
  const h16_t* biz = wih + (1 * DD + tn * 16 + m) * KX + hs * 16;
  const h16_t* bin = wih + (2 * DD + tn * 16 + m) * KX + hs * 16;
  const h16_t* bhr = whh + (0 * DD + tn * 16 + m) * DD + hs * 16;
  const h16_t* bhz = whh + (1 * DD + tn * 16 + m) * DD + hs * 16;
  const h16_t* bhn = whh + (2 * DD + tn * 16 + m) * DD + hs * 16;

  v8f air = {0.f,0.f,0.f,0.f,0.f,0.f,0.f,0.f}, aiz = air, ain = air;
  v8f ahr = air, ahz = air, ahn = air;
  for (int k = 0; k < KX; k += 32) {
    v16h avx = ldA(ax + k);
    air = wmma_step(avx, ldB(bir + k), air);
    aiz = wmma_step(avx, ldB(biz + k), aiz);
    ain = wmma_step(avx, ldB(bin + k), ain);
    v16h avh = ldA(ah + k);
    ahr = wmma_step(avh, ldB(bhr + k), ahr);
    ahz = wmma_step(avh, ldB(bhz + k), ahz);
    ahn = wmma_step(avh, ldB(bhn + k), ahn);
  }
  const int j = tn * 16 + m;
#pragma unroll
  for (int r = 0; r < 8; ++r) {
    const int b = tm * 16 + hs * 8 + r;
    float rg = 1.f / (1.f + __expf(-(air[r] + ahr[r])));
    float zg = 1.f / (1.f + __expf(-(aiz[r] + ahz[r])));
    float ng = tanhf(ain[r] + rg * ahn[r]);
    float hp = h32[b * DD + j];
    float hn = fmaxf((1.f - zg) * ng + zg * hp, 0.f);     // relu(dropout=0)
    h32[b * DD + j]   = hn;
    hnext[b * DD + j] = (h16_t)hn;
    dhac[(tstep * NB + b) * HK + EE + j] = (h16_t)hn;
  }
}

// ---------------------------------------------------------------------------
// Heads (after the recurrence): (64*32 x 96) = dhac(2048x768) @ Whead^T,
// fused bias + scatter to mel (B, NM, TD) and gate (B, TD, 1) layouts.
// ---------------------------------------------------------------------------
__global__ void k_heads(const h16_t* __restrict__ dhac, const h16_t* __restrict__ whead,
                        const float* __restrict__ b_mel, const float* __restrict__ b_gate,
                        float* __restrict__ out) {
  const int lane = threadIdx.x & 31;
  const int g = blockIdx.x * 4 + (threadIdx.x >> 5);      // 0..767
  const int tm = g / 6, tn = g % 6;
  v8f acc = wmma_mac16(dhac + tm * 16 * HK, HK, whead + tn * 16 * HK, HK, HK, lane);
  const int col = tn * 16 + (lane & 15);
  const int r0  = (lane >> 4) * 8;
#pragma unroll
  for (int r = 0; r < 8; ++r) {
    int row = tm * 16 + r0 + r;                           // row = t*NB + b
    int t = row >> 5, b = row & 31;
    if (col < NM)       out[b * (NM * TD) + col * TD + t] = acc[r] + b_mel[col];
    else if (col == NM) out[NB * NM * TD + b * TD + t]    = acc[r] + b_gate[0];
  }
}

// ---------------------------------------------------------------------------
extern "C" void kernel_launch(void* const* d_in, const int* in_sizes, int n_in,
                              void* d_out, int out_size, void* d_ws, size_t ws_size,
                              hipStream_t stream) {
  (void)in_sizes; (void)n_in; (void)out_size; (void)ws_size;
  const float* dec    = (const float*)d_in[0];
  const float* alignI = (const float*)d_in[1];
  const float* W_att  = (const float*)d_in[2];
  const float* W_ih   = (const float*)d_in[3];
  const float* W_hh   = (const float*)d_in[4];
  const float* W_mel  = (const float*)d_in[5];
  const float* b_mel  = (const float*)d_in[6];
  const float* W_gate = (const float*)d_in[7];
  const float* b_gate = (const float*)d_in[8];
  float* out = (float*)d_out;

  char* base = (char*)d_ws;
  size_t off = 0;
  auto alloc = [&](size_t bytes) -> void* {
    void* r = base + off;
    off = (off + bytes + 255) & ~(size_t)255;
    return r;
  };
  h16_t* wa16    = (h16_t*)alloc((size_t)EE * DD * 2);
  h16_t* wih16   = (h16_t*)alloc((size_t)G3 * KX * 2);
  h16_t* whh16   = (h16_t*)alloc((size_t)G3 * DD * 2);
  h16_t* whead16 = (h16_t*)alloc((size_t)96 * HK * 2);
  h16_t* seq16   = (h16_t*)alloc((size_t)TD * NB * EE * 2);
  h16_t* al16T   = (h16_t*)alloc((size_t)NB * EE * TE * 2);  // [b][e][t] f16
  h16_t* x16     = (h16_t*)alloc((size_t)NB * KX * 2);
  h16_t* hbuf0   = (h16_t*)alloc((size_t)NB * DD * 2);       // double-buffered h
  h16_t* hbuf1   = (h16_t*)alloc((size_t)NB * DD * 2);
  float* h32     = (float*)alloc((size_t)NB * DD * 4);
  h16_t* dhac16  = (h16_t*)alloc((size_t)TD * NB * HK * 2);

  k_convert<<<dim3(256), dim3(256), 0, stream>>>(W_att, W_ih, W_hh, W_mel, W_gate,
                                                 dec, alignI,
                                                 wa16, wih16, whh16, whead16,
                                                 seq16, al16T, h32, hbuf0);

  const size_t smem = (size_t)TE * ALD * 2 + 4u * TE * 4u;  // align cache + stats
  for (int t = 0; t < TD; ++t) {
    h16_t* hcur  = (t & 1) ? hbuf1 : hbuf0;
    h16_t* hnext = (t & 1) ? hbuf0 : hbuf1;
    k_attn<<<dim3(NB), dim3(256), smem, stream>>>(al16T, hcur, wa16, seq16,
                                                  x16, dhac16, t);
    k_gru <<<dim3(16), dim3(128), 0,    stream>>>(x16, hcur, wih16, whh16,
                                                  h32, hnext, dhac16, t);
  }
  k_heads<<<dim3(192), dim3(128), 0, stream>>>(dhac16, whead16, b_mel, b_gate, out);
}
